// HamiltonianPotentialNet_31473520345706
// MI455X (gfx1250) — matrix operand order repaired and checked by reading, not compiled, not run on hardware
//
#include <hip/hip_runtime.h>
#include <hip/hip_bf16.h>

typedef __attribute__((ext_vector_type(16))) _Float16 v16h;
typedef __attribute__((ext_vector_type(8)))  _Float16 v8h;
typedef __attribute__((ext_vector_type(8)))  float    v8f;

// ---------------------------------------------------------------------------
// WMMA GEMM:  Y[M x 128] = act( A[M x K] * W[K x 128] + bias ),  K = NSLAB*32.
// A is assembled from up to 3 K-segments of 128 (seg0 may be 32 for the
// encoders), each with its own base pointer and optional per-row gather index
// (folds [node[recv] | node[send] | edge] concat into the operand loader).
// W is pre-packed in per-lane B-fragment layout (pack_w_kernel).
// Block = 128 threads = 4 waves. Each wave computes 32 rows x 128 cols:
//   2 A-fragments, 16 f32 accumulator tiles (128 VGPRs), 2 WMMAs per B-frag.
// NSLAB is a template parameter and the K-loop is fully unrolled so the
// scheduler interleaves next-slab A loads with the current WMMA stream;
// B fragments use distance-2 pipelining (two loads in flight).
// Epilogue has an unguarded fast path for fully in-range wave tiles.
// ---------------------------------------------------------------------------
template <int NSLAB>
__global__ __launch_bounds__(128) void gemm_wmma_kernel(
    const _Float16* __restrict__ A0, const int* __restrict__ idx0, int ld0,
    const _Float16* __restrict__ A1, const int* __restrict__ idx1,
    const _Float16* __restrict__ A2, const int* __restrict__ idx2,
    const _Float16* __restrict__ Wf, const float* __restrict__ bias,
    _Float16* __restrict__ Y, int M, int do_relu)
{
    constexpr int NSEG = (NSLAB * 32 + 127) / 128;
    const int wave = threadIdx.x >> 5;
    const int lane = threadIdx.x & 31;
    const int hi   = lane >> 4;     // 0: lanes 0-15, 1: lanes 16-31
    const int l16  = lane & 15;
    const int rowBase = blockIdx.x * 128 + wave * 32;
    const int m0 = rowBase + l16;          // rows rowBase..rowBase+15
    const int m1 = m0 + 16;                // rows rowBase+16..rowBase+31
    const int mc0 = (m0 < M) ? m0 : (M - 1);   // clamp; stores are guarded
    const int mc1 = (m1 < M) ? m1 : (M - 1);

    // Hoisted per-segment row base pointers (gather resolved once).
    const _Float16 *p00, *p01, *p10 = nullptr, *p11 = nullptr,
                   *p20 = nullptr, *p21 = nullptr;
    {
        const int r0 = idx0 ? idx0[mc0] : mc0;
        const int r1 = idx0 ? idx0[mc1] : mc1;
        p00 = A0 + (size_t)r0 * ld0 + hi * 8;
        p01 = A0 + (size_t)r1 * ld0 + hi * 8;
    }
    if (NSEG > 1) {
        const int r0 = idx1 ? idx1[mc0] : mc0;
        const int r1 = idx1 ? idx1[mc1] : mc1;
        p10 = A1 + (size_t)r0 * 128 + hi * 8;
        p11 = A1 + (size_t)r1 * 128 + hi * 8;
    }
    if (NSEG > 2) {
        const int r0 = idx2 ? idx2[mc0] : mc0;
        const int r1 = idx2 ? idx2[mc1] : mc1;
        p20 = A2 + (size_t)r0 * 128 + hi * 8;
        p21 = A2 + (size_t)r1 * 128 + hi * 8;
    }

    v8f acc0[8] = {};
    v8f acc1[8] = {};

    #pragma unroll
    for (int s = 0; s < NSLAB; ++s) {
        const int kb = s << 5;                 // compile-time constant
        const _Float16 *q0, *q1;
        int kl;
        if (kb < 128)      { q0 = p00; q1 = p01; kl = kb; }
        else if (kb < 256) { q0 = p10; q1 = p11; kl = kb - 128; }
        else               { q0 = p20; q1 = p21; kl = kb - 256; }

        // A fragments (16-bit A layout, ISA 7.12.2): lane<16 holds its row at
        // K = kb+{0..7,16..23}; lanes>=16 at K = kb+8+{0..7,16..23}.
        const v8h x0 = *(const v8h*)(q0 + kl);
        const v8h x1 = *(const v8h*)(q0 + kl + 16);
        const v8h y0 = *(const v8h*)(q1 + kl);
        const v8h y1 = *(const v8h*)(q1 + kl + 16);
        v16h a0, a1;
        #pragma unroll
        for (int i = 0; i < 8; ++i) {
            a0[i] = x0[i]; a0[i + 8] = x1[i];
            a1[i] = y0[i]; a1[i + 8] = y1[i];
        }

        // B fragments: distance-2 software pipeline (two loads in flight).
        const _Float16* wp = Wf + (size_t)s * 4096 + lane * 16;
        v16h b0 = *(const v16h*)wp;
        v16h b1 = *(const v16h*)(wp + 512);
        #pragma unroll
        for (int nt = 0; nt < 8; ++nt) {
            const v16h b2 = (nt < 6) ? *(const v16h*)(wp + (nt + 2) * 512) : b0;
            acc0[nt] = __builtin_amdgcn_wmma_f32_16x16x32_f16(
                false, a0, false, b0, (short)0, acc0[nt], false, false);
            acc1[nt] = __builtin_amdgcn_wmma_f32_16x16x32_f16(
                false, a1, false, b0, (short)0, acc1[nt], false, false);
            b0 = b1; b1 = b2;
        }
    }

    // C/D layout: VGPR r -> M = r + 8*hi ; N = l16 (per 16-col tile nt).
    if (rowBase + 31 < M) {
        // Fast path: whole wave tile in range -> unguarded b16 stores.
        #pragma unroll
        for (int nt = 0; nt < 8; ++nt) {
            const int col = nt * 16 + l16;
            const float bv = bias[col];
            _Float16* yp = Y + (size_t)(rowBase + hi * 8) * 128 + col;
            #pragma unroll
            for (int r = 0; r < 8; ++r) {
                float v0 = acc0[nt][r] + bv;
                float v1 = acc1[nt][r] + bv;
                if (do_relu) {
                    v0 = v0 > 0.0f ? v0 : 0.0f;
                    v1 = v1 > 0.0f ? v1 : 0.0f;
                }
                yp[(size_t)r * 128]        = (_Float16)v0;
                yp[(size_t)(r + 16) * 128] = (_Float16)v1;
            }
        }
    } else {
        #pragma unroll
        for (int nt = 0; nt < 8; ++nt) {
            const int col = nt * 16 + l16;
            const float bv = bias[col];
            #pragma unroll
            for (int r = 0; r < 8; ++r) {
                const int rr0 = rowBase + r + hi * 8;
                if (rr0 < M) {
                    float v = acc0[nt][r] + bv;
                    if (do_relu) v = v > 0.0f ? v : 0.0f;
                    Y[(size_t)rr0 * 128 + col] = (_Float16)v;
                }
                const int rr1 = rr0 + 16;
                if (rr1 < M) {
                    float v = acc1[nt][r] + bv;
                    if (do_relu) v = v > 0.0f ? v : 0.0f;
                    Y[(size_t)rr1 * 128 + col] = (_Float16)v;
                }
            }
        }
    }
}

// Pack W[K x 128] (f32 row-major, zero-padded to Kpad) into the per-lane
// B-fragment layout consumed above:
//   t = s*4096 + nt*512 + lane*16 + h  ->  k = s*32 + 8*(lane>>4) + (h<8 ? h : h+8),
//                                          n = nt*16 + (lane&15)
__global__ void pack_w_kernel(const float* __restrict__ W, int Kreal, int total,
                              _Float16* __restrict__ Wf)
{
    int t = blockIdx.x * 256 + threadIdx.x;
    if (t >= total) return;
    int h    = t & 15;
    int lane = (t >> 4) & 31;
    int nt   = (t >> 9) & 7;
    int s    = t >> 12;
    int k = (s << 5) + ((lane >> 4) << 3) + (h < 8 ? h : h + 8);
    int n = (nt << 4) + (lane & 15);
    Wf[t] = (_Float16)((k < Kreal) ? W[k * 128 + n] : 0.0f);
}

// Node features, padded K=23 -> 32, converted to f16.
__global__ void node_feat_kernel(const float* __restrict__ x, const float* __restrict__ v,
                                 const float* __restrict__ rho, const int* __restrict__ pt,
                                 const float* __restrict__ temb,
                                 _Float16* __restrict__ out, int N)
{
    int t = blockIdx.x * 256 + threadIdx.x;
    if (t >= N * 32) return;
    int n = t >> 5, j = t & 31;
    float val = 0.0f;
    if (j < 3)       val = x[n * 3 + j];
    else if (j < 6)  val = v[n * 3 + (j - 3)];
    else if (j == 6) val = rho[n];
    else if (j < 23) val = temb[pt[n] * 16 + (j - 7)];
    out[t] = (_Float16)val;
}

// Edge features, padded K=5 -> 32.
__global__ void edge_feat_kernel(const float* __restrict__ ef, _Float16* __restrict__ out, int E)
{
    int t = blockIdx.x * 256 + threadIdx.x;
    if (t >= E * 32) return;
    int e = t >> 5, j = t & 31;
    out[t] = (_Float16)((j < 5) ? ef[e * 5 + j] : 0.0f);
}

// LayerNorm over 128 features; optional residual master += LN(H); optional
// separate LN output (needed when edge_upd feeds both residual and segment sum).
__global__ __launch_bounds__(128) void ln_kernel(
    const _Float16* __restrict__ H, const float* __restrict__ g, const float* __restrict__ b,
    _Float16* __restrict__ master, _Float16* __restrict__ out, int M)
{
    __shared__ float red[128];
    const int row = blockIdx.x;
    const int j = threadIdx.x;
    const size_t o = (size_t)row * 128 + j;
    const float xv = (float)H[o];
    red[j] = xv; __syncthreads();
    #pragma unroll
    for (int s = 64; s > 0; s >>= 1) { if (j < s) red[j] += red[j + s]; __syncthreads(); }
    const float mean = red[0] * (1.0f / 128.0f);
    __syncthreads();
    const float d = xv - mean;
    red[j] = d * d; __syncthreads();
    #pragma unroll
    for (int s = 64; s > 0; s >>= 1) { if (j < s) red[j] += red[j + s]; __syncthreads(); }
    const float var = red[0] * (1.0f / 128.0f);
    const float y = d * rsqrtf(var + 1e-5f) * g[j] + b[j];
    if (out) out[o] = (_Float16)y;
    if (master) master[o] = (_Float16)((float)master[o] + y);
}

__global__ void zero_f32_kernel(float* __restrict__ p, int n)
{
    int t = blockIdx.x * 256 + threadIdx.x;
    if (t < n) p[t] = 0.0f;
}

// segment_sum: agg[recv[e]] += edge_upd[e]   (global_atomic_add_f32)
__global__ void scatter_kernel(const _Float16* __restrict__ eupd, const int* __restrict__ recv,
                               float* __restrict__ agg, long total)
{
    long t = (long)blockIdx.x * 256 + threadIdx.x;
    if (t >= total) return;
    int e = (int)(t >> 7), j = (int)(t & 127);
    unsafeAtomicAdd(&agg[(size_t)recv[e] * 128 + j], (float)eupd[t]);
}

__global__ void cvt_f32_f16_kernel(const float* __restrict__ src, _Float16* __restrict__ dst, int n)
{
    int t = blockIdx.x * 256 + threadIdx.x;
    if (t < n) dst[t] = (_Float16)src[t];
}

// Final decoder layer: [N x 128] . [128 x 1] + b -> f32 output.
__global__ void dec_out_kernel(const _Float16* __restrict__ H, const float* __restrict__ w,
                               const float* __restrict__ b, float* __restrict__ out, int N)
{
    int n = blockIdx.x * 256 + threadIdx.x;
    if (n >= N) return;
    float s = b[0];
    for (int k = 0; k < 128; ++k) s += (float)H[(size_t)n * 128 + k] * w[k];
    out[n] = s;
}

// ---------------------------------------------------------------------------
extern "C" void kernel_launch(void* const* d_in, const int* in_sizes, int n_in,
                              void* d_out, int out_size, void* d_ws, size_t ws_size,
                              hipStream_t stream)
{
    const float* x     = (const float*)d_in[0];
    const float* vel   = (const float*)d_in[1];
    const float* rho   = (const float*)d_in[2];
    const int*   ptype = (const int*)d_in[3];
    const int*   eidx  = (const int*)d_in[4];
    const float* efeat = (const float*)d_in[5];

    const int N = in_sizes[2];
    const int E = in_sizes[4] / 2;
    const int* senders   = eidx;        // edge_index[0]
    const int* receivers = eidx + E;    // edge_index[1]
    const int NP = (N + 63) & ~63;

    // ---- params: jax pytree flatten, dicts in sorted-key order ----
    int p = 6;
    auto F = [&](int& i) { return (const float*)d_in[i++]; };
    // decoder: [{b,w} x 3]
    const float* dec_b0 = F(p); const float* dec_w0 = F(p);
    const float* dec_b1 = F(p); const float* dec_w1 = F(p);
    const float* dec_b2 = F(p); const float* dec_w2 = F(p);
    // edge_enc: ln{b,g}, mlp[{b,w} x 3]
    const float* ee_lnb = F(p); const float* ee_lng = F(p);
    const float* ee_b0 = F(p); const float* ee_w0 = F(p);
    const float* ee_b1 = F(p); const float* ee_w1 = F(p);
    const float* ee_b2 = F(p); const float* ee_w2 = F(p);
    // node_enc
    const float* ne_lnb = F(p); const float* ne_lng = F(p);
    const float* ne_b0 = F(p); const float* ne_w0 = F(p);
    const float* ne_b1 = F(p); const float* ne_w1 = F(p);
    const float* ne_b2 = F(p); const float* ne_w2 = F(p);
    // steps[4]: edge_fn{ln{b,g}, mlp x3}, node_fn{ln{b,g}, mlp x3}
    struct StepP {
        const float *elnb, *elng, *eb0, *ew0, *eb1, *ew1, *eb2, *ew2;
        const float *nlnb, *nlng, *nb0, *nw0, *nb1, *nw1, *nb2, *nw2;
    } st[4];
    for (int s = 0; s < 4; ++s) {
        st[s].elnb = F(p); st[s].elng = F(p);
        st[s].eb0 = F(p); st[s].ew0 = F(p);
        st[s].eb1 = F(p); st[s].ew1 = F(p);
        st[s].eb2 = F(p); st[s].ew2 = F(p);
        st[s].nlnb = F(p); st[s].nlng = F(p);
        st[s].nb0 = F(p); st[s].nw0 = F(p);
        st[s].nb1 = F(p); st[s].nw1 = F(p);
        st[s].nb2 = F(p); st[s].nw2 = F(p);
    }
    const float* type_emb = F(p);

    // ---- workspace carve ----
    size_t off = 0; char* base = (char*)d_ws;
    auto alloc = [&](size_t bytes) -> void* {
        off = (off + 255) & ~(size_t)255;
        void* r = base + off; off += bytes; return r;
    };
    _Float16* node16  = (_Float16*)alloc((size_t)NP * 128 * 2);
    _Float16* edge16  = (_Float16*)alloc((size_t)E * 128 * 2);
    _Float16* h1      = (_Float16*)alloc((size_t)E * 128 * 2);
    _Float16* h2      = (_Float16*)alloc((size_t)E * 128 * 2);   // also reused as edge_upd
    _Float16* nfeat   = (_Float16*)alloc((size_t)NP * 32 * 2);
    _Float16* efeat16 = (_Float16*)alloc((size_t)E * 32 * 2);
    float*    agg     = (float*)alloc((size_t)NP * 128 * 4);
    _Float16* agg16   = (_Float16*)alloc((size_t)NP * 128 * 2);

    auto packW = [&](const float* src, int Kreal, int Kpad) -> _Float16* {
        _Float16* dst = (_Float16*)alloc((size_t)Kpad * 128 * 2);
        int total = Kpad * 128;
        pack_w_kernel<<<(total + 255) / 256, 256, 0, stream>>>(src, Kreal, total, dst);
        return dst;
    };
    _Float16* Wne0 = packW(ne_w0, 23, 32);
    _Float16* Wne1 = packW(ne_w1, 128, 128);
    _Float16* Wne2 = packW(ne_w2, 128, 128);
    _Float16* Wee0 = packW(ee_w0, 5, 32);
    _Float16* Wee1 = packW(ee_w1, 128, 128);
    _Float16* Wee2 = packW(ee_w2, 128, 128);
    _Float16 *We0[4], *We1[4], *We2[4], *Wn0[4], *Wn1[4], *Wn2[4];
    for (int s = 0; s < 4; ++s) {
        We0[s] = packW(st[s].ew0, 384, 384);
        We1[s] = packW(st[s].ew1, 128, 128);
        We2[s] = packW(st[s].ew2, 128, 128);
        Wn0[s] = packW(st[s].nw0, 256, 256);
        Wn1[s] = packW(st[s].nw1, 128, 128);
        Wn2[s] = packW(st[s].nw2, 128, 128);
    }
    _Float16* Wd0 = packW(dec_w0, 128, 128);
    _Float16* Wd1 = packW(dec_w1, 128, 128);

    auto gemm = [&](const _Float16* A0, const int* i0, int ld0,
                    const _Float16* A1, const int* i1,
                    const _Float16* A2, const int* i2,
                    int K, const _Float16* Wf, const float* bias,
                    _Float16* Y, int M, int relu) {
        dim3 grid((M + 127) / 128);
        switch (K) {
        case 32:
            gemm_wmma_kernel<1><<<grid, 128, 0, stream>>>(
                A0, i0, ld0, A1, i1, A2, i2, Wf, bias, Y, M, relu);
            break;
        case 128:
            gemm_wmma_kernel<4><<<grid, 128, 0, stream>>>(
                A0, i0, ld0, A1, i1, A2, i2, Wf, bias, Y, M, relu);
            break;
        case 256:
            gemm_wmma_kernel<8><<<grid, 128, 0, stream>>>(
                A0, i0, ld0, A1, i1, A2, i2, Wf, bias, Y, M, relu);
            break;
        default:
            gemm_wmma_kernel<12><<<grid, 128, 0, stream>>>(
                A0, i0, ld0, A1, i1, A2, i2, Wf, bias, Y, M, relu);
            break;
        }
    };

    // ---- encoders ----
    node_feat_kernel<<<(N * 32 + 255) / 256, 256, 0, stream>>>(x, vel, rho, ptype, type_emb, nfeat, N);
    gemm(nfeat, nullptr, 32, nullptr, nullptr, nullptr, nullptr, 32, Wne0, ne_b0, h1, N, 1);
    gemm(h1, nullptr, 128, nullptr, nullptr, nullptr, nullptr, 128, Wne1, ne_b1, h2, N, 1);
    gemm(h2, nullptr, 128, nullptr, nullptr, nullptr, nullptr, 128, Wne2, ne_b2, h1, N, 0);
    ln_kernel<<<N, 128, 0, stream>>>(h1, ne_lng, ne_lnb, nullptr, node16, N);

    edge_feat_kernel<<<(E * 32 + 255) / 256, 256, 0, stream>>>(efeat, efeat16, E);
    gemm(efeat16, nullptr, 32, nullptr, nullptr, nullptr, nullptr, 32, Wee0, ee_b0, h1, E, 1);
    gemm(h1, nullptr, 128, nullptr, nullptr, nullptr, nullptr, 128, Wee1, ee_b1, h2, E, 1);
    gemm(h2, nullptr, 128, nullptr, nullptr, nullptr, nullptr, 128, Wee2, ee_b2, h1, E, 0);
    ln_kernel<<<E, 128, 0, stream>>>(h1, ee_lng, ee_lnb, nullptr, edge16, E);

    // ---- message passing steps ----
    for (int s = 0; s < 4; ++s) {
        // edge_fn on concat [node[recv] | node[send] | edge] via 3-segment A
        gemm(node16, receivers, 128, node16, senders, edge16, nullptr,
             384, We0[s], st[s].eb0, h1, E, 1);
        gemm(h1, nullptr, 128, nullptr, nullptr, nullptr, nullptr, 128, We1[s], st[s].eb1, h2, E, 1);
        gemm(h2, nullptr, 128, nullptr, nullptr, nullptr, nullptr, 128, We2[s], st[s].eb2, h1, E, 0);
        // edge += LN(h1); h2 = LN(h1) (= edge_upd, feeds segment_sum)
        ln_kernel<<<E, 128, 0, stream>>>(h1, st[s].elng, st[s].elnb, edge16, h2, E);

        zero_f32_kernel<<<(NP * 128 + 255) / 256, 256, 0, stream>>>(agg, NP * 128);
        scatter_kernel<<<(int)(((long)E * 128 + 255) / 256), 256, 0, stream>>>(
            h2, receivers, agg, (long)E * 128);
        cvt_f32_f16_kernel<<<(NP * 128 + 255) / 256, 256, 0, stream>>>(agg, agg16, NP * 128);

        // node_fn on concat [node | agg] via 2-segment A
        gemm(node16, nullptr, 128, agg16, nullptr, nullptr, nullptr,
             256, Wn0[s], st[s].nb0, h1, N, 1);
        gemm(h1, nullptr, 128, nullptr, nullptr, nullptr, nullptr, 128, Wn1[s], st[s].nb1, h2, N, 1);
        gemm(h2, nullptr, 128, nullptr, nullptr, nullptr, nullptr, 128, Wn2[s], st[s].nb2, h1, N, 0);
        ln_kernel<<<N, 128, 0, stream>>>(h1, st[s].nlng, st[s].nlnb, node16, nullptr, N);
    }

    // ---- decoder ----
    gemm(node16, nullptr, 128, nullptr, nullptr, nullptr, nullptr, 128, Wd0, dec_b0, h1, N, 1);
    gemm(h1, nullptr, 128, nullptr, nullptr, nullptr, nullptr, 128, Wd1, dec_b1, h2, N, 1);
    dec_out_kernel<<<(N + 255) / 256, 256, 0, stream>>>(h2, dec_w2, dec_b2, (float*)d_out, N);
}
